// SAGE_40132174414141
// MI455X (gfx1250) — matrix operand order repaired and checked by reading, not compiled, not run on hardware
//
#include <hip/hip_runtime.h>
#include <hip/hip_bf16.h>

typedef __attribute__((ext_vector_type(16))) __bf16          v16bf;
typedef __attribute__((ext_vector_type(8)))  float           v8f;
typedef __attribute__((ext_vector_type(16))) unsigned short  us16;
typedef __attribute__((ext_vector_type(8)))  unsigned short  us8;

// float -> bf16 using the hardware convert (v_cvt_*bf16_f32 class ops)
__device__ __forceinline__ __bf16 f2bf(float f) { return (__bf16)f; }
__device__ __forceinline__ unsigned short f2bfu(float f) {
  return __builtin_bit_cast(unsigned short, (__bf16)f);
}

// Build A fragment (16x32 bf16, ISA 7.12.2 layout) for this lane from a
// row of fp32 data, scaled. k0 = kb*32 + (lane>=16 ? 8 : 0).
// elements 0..7  <- K = k0 + e
// elements 8..15 <- K = k0 + 16 + (e-8)
__device__ __forceinline__ v16bf pack_row(const float* __restrict__ p, int k0, float scale) {
  const float4 f0 = *reinterpret_cast<const float4*>(p + k0);
  const float4 f1 = *reinterpret_cast<const float4*>(p + k0 + 4);
  const float4 f2 = *reinterpret_cast<const float4*>(p + k0 + 16);
  const float4 f3 = *reinterpret_cast<const float4*>(p + k0 + 20);
  v16bf t;
  t[0]  = f2bf(f0.x * scale); t[1]  = f2bf(f0.y * scale);
  t[2]  = f2bf(f0.z * scale); t[3]  = f2bf(f0.w * scale);
  t[4]  = f2bf(f1.x * scale); t[5]  = f2bf(f1.y * scale);
  t[6]  = f2bf(f1.z * scale); t[7]  = f2bf(f1.w * scale);
  t[8]  = f2bf(f2.x * scale); t[9]  = f2bf(f2.y * scale);
  t[10] = f2bf(f2.z * scale); t[11] = f2bf(f2.w * scale);
  t[12] = f2bf(f3.x * scale); t[13] = f2bf(f3.y * scale);
  t[14] = f2bf(f3.z * scale); t[15] = f2bf(f3.w * scale);
  return t;
}

// Same but element-wise product of two fp32 rows (predictor z = h[u]*h[v]).
__device__ __forceinline__ v16bf pack_prod(const float* __restrict__ pa,
                                           const float* __restrict__ pb, int k0) {
  const float4 a0 = *reinterpret_cast<const float4*>(pa + k0);
  const float4 a1 = *reinterpret_cast<const float4*>(pa + k0 + 4);
  const float4 a2 = *reinterpret_cast<const float4*>(pa + k0 + 16);
  const float4 a3 = *reinterpret_cast<const float4*>(pa + k0 + 20);
  const float4 b0 = *reinterpret_cast<const float4*>(pb + k0);
  const float4 b1 = *reinterpret_cast<const float4*>(pb + k0 + 4);
  const float4 b2 = *reinterpret_cast<const float4*>(pb + k0 + 16);
  const float4 b3 = *reinterpret_cast<const float4*>(pb + k0 + 20);
  v16bf t;
  t[0]  = f2bf(a0.x * b0.x); t[1]  = f2bf(a0.y * b0.y);
  t[2]  = f2bf(a0.z * b0.z); t[3]  = f2bf(a0.w * b0.w);
  t[4]  = f2bf(a1.x * b1.x); t[5]  = f2bf(a1.y * b1.y);
  t[6]  = f2bf(a1.z * b1.z); t[7]  = f2bf(a1.w * b1.w);
  t[8]  = f2bf(a2.x * b2.x); t[9]  = f2bf(a2.y * b2.y);
  t[10] = f2bf(a2.z * b2.z); t[11] = f2bf(a2.w * b2.w);
  t[12] = f2bf(a3.x * b3.x); t[13] = f2bf(a3.y * b3.y);
  t[14] = f2bf(a3.z * b3.z); t[15] = f2bf(a3.w * b3.w);
  return t;
}

// ---------------------------------------------------------------------------
// Pre-swizzle a 128x128 fp32 weight matrix (row = K, col = N) into bf16
// WMMA B-fragment order. Tile (kb,nt) covers K=[kb*32,+32), N=[nt*16,+16).
// Lane L of a tile stores 16 bf16: col = nt*16 + (L&15),
// K = kb*32 + (L>=16 ? 8 : 0) + (e<8 ? e : e+8). grid = 32 tiles, block = 32 lanes.
// ---------------------------------------------------------------------------
__global__ __launch_bounds__(32) void swizzle_w_kernel(const float* __restrict__ W,
                                                       unsigned short* __restrict__ dst) {
  const int tile = blockIdx.x;          // 0..31 : (kb*8 + nt)
  const int lane = threadIdx.x;         // 0..31
  const int kb   = (tile >> 3) * 32;
  const int nt   = tile & 7;
  const int col  = nt * 16 + (lane & 15);
  const int kbase = kb + ((lane & 16) ? 8 : 0);
  us16 t;
#pragma unroll
  for (int e = 0; e < 16; ++e) t[e] = f2bfu(W[(kbase + (e < 8 ? e : e + 8)) * 128 + col]);
  *reinterpret_cast<us16*>(dst + (size_t)tile * 512 + lane * 16) = t;
}

__global__ void zero_kernel(float* __restrict__ p, long long n) {
  long long i = (long long)blockIdx.x * blockDim.x + threadIdx.x;
  if (i < n) p[i] = 0.0f;
}

// ---------------------------------------------------------------------------
// Weighted scatter-add: agg[dst] += w * h[src], cnt[dst] += 1.
// One thread per (edge, 4-feature chunk): 32 threads per edge.
// ---------------------------------------------------------------------------
__global__ __launch_bounds__(256) void scatter_kernel(
    const float* __restrict__ h, const int* __restrict__ src,
    const int* __restrict__ dst, const float* __restrict__ w,
    float* __restrict__ agg, float* __restrict__ cnt, int E) {
  long long tid = (long long)blockIdx.x * blockDim.x + threadIdx.x;
  if (tid >= (long long)E * 32) return;
  const int e = (int)(tid >> 5);
  const int c = (int)(tid & 31);
  const int s = src[e];
  const int d = dst[e];
  const float wt = w[e];
  const float4 hv = *reinterpret_cast<const float4*>(h + (size_t)s * 128 + c * 4);
  float* ap = agg + (size_t)d * 128 + c * 4;
  atomicAdd(ap + 0, hv.x * wt);
  atomicAdd(ap + 1, hv.y * wt);
  atomicAdd(ap + 2, hv.z * wt);
  atomicAdd(ap + 3, hv.w * wt);
  if (c == 0) atomicAdd(cnt + d, 1.0f);
}

// ---------------------------------------------------------------------------
// Fused SAGE layer: out = [relu]( h@Wself + (agg/max(cnt,1))@Wneigh + b ).
// Block = 128 threads = 4 waves; each wave computes 16 rows x 128 cols
// with 8 fp32 C fragments; K-loop = 4 steps of v_wmma_f32_16x16x32_bf16 x16.
// ---------------------------------------------------------------------------
__global__ __launch_bounds__(128) void sage_layer_kernel(
    const float* __restrict__ hin, const float* __restrict__ agg,
    const float* __restrict__ cnt, const unsigned short* __restrict__ wself,
    const unsigned short* __restrict__ wneigh, const float* __restrict__ bias,
    float* __restrict__ hout, int nrows, int relu_flag) {
  const int lane = threadIdx.x & 31;
  const int wave = threadIdx.x >> 5;
  const int l16  = lane & 15;
  const int hi   = lane >> 4;                 // 0 or 1
  const int rowBase = blockIdx.x * 64 + wave * 16;
  int r = rowBase + l16;
  if (r > nrows - 1) r = nrows - 1;           // clamp so EXEC stays all-ones
  const float cn  = cnt[r];
  // single v_rcp_f32 instead of the full IEEE divide expansion; neighbor
  // features already flow through bf16, so 1-ulp rcp is within budget.
  const float inv = __builtin_amdgcn_rcpf(fmaxf(cn, 1.0f));
  const float* hrow = hin + (size_t)r * 128;
  const float* arow = agg + (size_t)r * 128;
  const int klo = hi * 8;

  v8f acc[8];
#pragma unroll
  for (int nt = 0; nt < 8; ++nt)
#pragma unroll
    for (int j = 0; j < 8; ++j) acc[nt][j] = 0.0f;

#pragma unroll
  for (int kb = 0; kb < 4; ++kb) {
    const int k0 = kb * 32 + klo;
    const v16bf a1 = pack_row(hrow, k0, 1.0f);
    const v16bf a2 = pack_row(arow, k0, inv);
#pragma unroll
    for (int nt = 0; nt < 8; ++nt) {
      const v16bf bS = *reinterpret_cast<const v16bf*>(
          wself + (size_t)(kb * 8 + nt) * 512 + lane * 16);
      acc[nt] = __builtin_amdgcn_wmma_f32_16x16x32_bf16(
          false, a1, false, bS, (short)0, acc[nt], false, false);
      const v16bf bN = *reinterpret_cast<const v16bf*>(
          wneigh + (size_t)(kb * 8 + nt) * 512 + lane * 16);
      acc[nt] = __builtin_amdgcn_wmma_f32_16x16x32_bf16(
          false, a2, false, bN, (short)0, acc[nt], false, false);
    }
  }

#pragma unroll
  for (int nt = 0; nt < 8; ++nt) {
    const int col = nt * 16 + l16;
    const float bb = bias[col];
#pragma unroll
    for (int j = 0; j < 8; ++j) {
      const int row = rowBase + hi * 8 + j;   // C layout: lanes>=16 hold M=j+8
      float v = acc[nt][j] + bb;
      if (relu_flag) v = fmaxf(v, 0.0f);
      if (row < nrows) hout[(size_t)row * 128 + col] = v;
    }
  }
}

// ---------------------------------------------------------------------------
// Fused link predictor: z = h[u]*h[v]; relu(z@pw1+pb1); relu(.@pw2+pb2);
// out = .@pw3 + pb3. Each wave owns 16 pairs. GEMM1 output is staged to LDS
// in bf16 and reloaded as A fragments for GEMM2; final 128->1 dot is an
// LDS cross-lane reduction.
// ---------------------------------------------------------------------------
__global__ __launch_bounds__(128) void predictor_kernel(
    const float* __restrict__ h, const int* __restrict__ isrc,
    const int* __restrict__ idst, const unsigned short* __restrict__ w1z,
    const float* __restrict__ pb1, const unsigned short* __restrict__ w2z,
    const float* __restrict__ pb2, const float* __restrict__ pw3,
    const float* __restrict__ pb3, float* __restrict__ out, int P) {
  __shared__ unsigned short zbuf[4][16 * 128];  // per-wave bf16 z1 (16 rows)
  __shared__ float          pbuf[4][16 * 16];   // per-wave dot partials

  const int lane = threadIdx.x & 31;
  const int wave = threadIdx.x >> 5;
  const int l16  = lane & 15;
  const int hi   = lane >> 4;
  const int rowBase = blockIdx.x * 64 + wave * 16;
  int p = rowBase + l16;
  if (p > P - 1) p = P - 1;
  const float* hu = h + (size_t)isrc[p] * 128;
  const float* hv = h + (size_t)idst[p] * 128;
  const int klo = hi * 8;

  // ---- GEMM1: z1 = relu( (h[u]*h[v]) @ pw1 + pb1 ) ----
  v8f acc[8];
#pragma unroll
  for (int nt = 0; nt < 8; ++nt)
#pragma unroll
    for (int j = 0; j < 8; ++j) acc[nt][j] = 0.0f;

#pragma unroll
  for (int kb = 0; kb < 4; ++kb) {
    const v16bf a = pack_prod(hu, hv, kb * 32 + klo);
#pragma unroll
    for (int nt = 0; nt < 8; ++nt) {
      const v16bf b = *reinterpret_cast<const v16bf*>(
          w1z + (size_t)(kb * 8 + nt) * 512 + lane * 16);
      acc[nt] = __builtin_amdgcn_wmma_f32_16x16x32_bf16(
          false, a, false, b, (short)0, acc[nt], false, false);
    }
  }

  unsigned short* zme = zbuf[wave];
#pragma unroll
  for (int nt = 0; nt < 8; ++nt) {
    const int col = nt * 16 + l16;
    const float bb = pb1[col];
#pragma unroll
    for (int j = 0; j < 8; ++j) {
      const float v = fmaxf(acc[nt][j] + bb, 0.0f);
      zme[(hi * 8 + j) * 128 + col] = f2bfu(v);
    }
  }
  __syncthreads();

  // ---- GEMM2: z2 = relu( z1 @ pw2 + pb2 ) ----
  v8f acc2[8];
#pragma unroll
  for (int nt = 0; nt < 8; ++nt)
#pragma unroll
    for (int j = 0; j < 8; ++j) acc2[nt][j] = 0.0f;

  const unsigned short* zrow = zme + l16 * 128;
#pragma unroll
  for (int kb = 0; kb < 4; ++kb) {
    const int k0 = kb * 32 + klo;
    const us8 lo  = *reinterpret_cast<const us8*>(zrow + k0);
    const us8 hi8 = *reinterpret_cast<const us8*>(zrow + k0 + 16);
    const v16bf a = __builtin_bit_cast(
        v16bf, __builtin_shufflevector(lo, hi8, 0, 1, 2, 3, 4, 5, 6, 7, 8, 9,
                                       10, 11, 12, 13, 14, 15));
#pragma unroll
    for (int nt = 0; nt < 8; ++nt) {
      const v16bf b = *reinterpret_cast<const v16bf*>(
          w2z + (size_t)(kb * 8 + nt) * 512 + lane * 16);
      acc2[nt] = __builtin_amdgcn_wmma_f32_16x16x32_bf16(
          false, a, false, b, (short)0, acc2[nt], false, false);
    }
  }

  // ---- fused 128->1 dot with pw3 ----
  float part[8];
#pragma unroll
  for (int j = 0; j < 8; ++j) part[j] = 0.0f;
#pragma unroll
  for (int nt = 0; nt < 8; ++nt) {
    const int col = nt * 16 + l16;
    const float bb = pb2[col];
    const float wv = pw3[col];
#pragma unroll
    for (int j = 0; j < 8; ++j) {
      const float v = fmaxf(acc2[nt][j] + bb, 0.0f);
      part[j] += v * wv;
    }
  }
  float* pme = pbuf[wave];
#pragma unroll
  for (int j = 0; j < 8; ++j) pme[(hi * 8 + j) * 16 + l16] = part[j];
  __syncthreads();

  const float* prow = pme + l16 * 16;
  const float4 q0 = *reinterpret_cast<const float4*>(prow + 0);
  const float4 q1 = *reinterpret_cast<const float4*>(prow + 4);
  const float4 q2 = *reinterpret_cast<const float4*>(prow + 8);
  const float4 q3 = *reinterpret_cast<const float4*>(prow + 12);
  const float s = pb3[0] + q0.x + q0.y + q0.z + q0.w + q1.x + q1.y + q1.z +
                  q1.w + q2.x + q2.y + q2.z + q2.w + q3.x + q3.y + q3.z + q3.w;
  const int pp = rowBase + l16;
  if (hi == 0 && pp < P) out[pp] = s;
}

// ---------------------------------------------------------------------------
extern "C" void kernel_launch(void* const* d_in, const int* in_sizes, int n_in,
                              void* d_out, int out_size, void* d_ws, size_t ws_size,
                              hipStream_t stream) {
  const float* x       = (const float*)d_in[0];
  const int*   srcs[3] = {(const int*)d_in[1], (const int*)d_in[4], (const int*)d_in[7]};
  const int*   dsts[3] = {(const int*)d_in[2], (const int*)d_in[5], (const int*)d_in[8]};
  const float* ws_e[3] = {(const float*)d_in[3], (const float*)d_in[6], (const float*)d_in[9]};
  const int*   pos_src = (const int*)d_in[10];
  const int*   pos_dst = (const int*)d_in[11];
  const int*   neg_src = (const int*)d_in[12];
  const int*   neg_dst = (const int*)d_in[13];
  const float* Wself[3]  = {(const float*)d_in[14], (const float*)d_in[17], (const float*)d_in[20]};
  const float* Wneigh[3] = {(const float*)d_in[15], (const float*)d_in[18], (const float*)d_in[21]};
  const float* bias[3]   = {(const float*)d_in[16], (const float*)d_in[19], (const float*)d_in[22]};
  const float* pw1 = (const float*)d_in[23];
  const float* pb1 = (const float*)d_in[24];
  const float* pw2 = (const float*)d_in[25];
  const float* pb2 = (const float*)d_in[26];
  const float* pw3 = (const float*)d_in[27];
  const float* pb3 = (const float*)d_in[28];

  const int N = in_sizes[0] / 128;   // 100000
  const int E = in_sizes[1];         // 600000
  const int P = in_sizes[10];        // 100000

  // workspace layout: [agg N*128 f32][cnt N f32][hA N*128 f32][wz 8*16384 bf16]
  float* agg = (float*)d_ws;
  float* cnt = agg + (size_t)N * 128;
  float* hA  = cnt + N;
  unsigned short* wz = (unsigned short*)(hA + (size_t)N * 128);

  // pre-swizzle all 8 weight matrices into WMMA B-fragment bf16 order
  const float* mats[8] = {Wself[0], Wneigh[0], Wself[1], Wneigh[1],
                          Wself[2], Wneigh[2], pw1, pw2};
  for (int m = 0; m < 8; ++m)
    swizzle_w_kernel<<<32, 32, 0, stream>>>(mats[m], wz + (size_t)m * 16384);

  float* out_pos = (float*)d_out;
  float* out_neg = out_pos + P;
  float* hOut    = out_pos + 2 * (size_t)P;   // final node embeddings [N,128]

  const float* hin[3]  = {x, hA, hOut};
  float*       hout[3] = {hA, hOut, hOut};    // layer2 runs in place (row-local)

  const long long zero_n = (long long)N * 129;           // agg + cnt contiguous
  const int zero_blocks  = (int)((zero_n + 255) / 256);
  const long long sc_n   = (long long)E * 32;
  const int sc_blocks    = (int)((sc_n + 255) / 256);
  const int layer_blocks = (N + 63) / 64;

  for (int l = 0; l < 3; ++l) {
    zero_kernel<<<zero_blocks, 256, 0, stream>>>(agg, zero_n);
    scatter_kernel<<<sc_blocks, 256, 0, stream>>>(hin[l], srcs[l], dsts[l],
                                                  ws_e[l], agg, cnt, E);
    sage_layer_kernel<<<layer_blocks, 128, 0, stream>>>(
        hin[l], agg, cnt, wz + (size_t)(2 * l) * 16384,
        wz + (size_t)(2 * l + 1) * 16384, bias[l], hout[l], N,
        (l < 2) ? 1 : 0);
  }

  const int pred_blocks = (P + 63) / 64;
  predictor_kernel<<<pred_blocks, 128, 0, stream>>>(
      hOut, pos_src, pos_dst, wz + 6 * 16384, pb1, wz + 7 * 16384, pb2, pw3,
      pb3, out_pos, P);
  predictor_kernel<<<pred_blocks, 128, 0, stream>>>(
      hOut, neg_src, neg_dst, wz + 6 * 16384, pb1, wz + 7 * 16384, pb2, pw3,
      pb3, out_neg, P);
}
